// DeformableTransformerDecoderLayer_34772055228676
// MI455X (gfx1250) — matrix operand order repaired and checked by reading, not compiled, not run on hardware
//
#include <hip/hip_runtime.h>

// ---------------------------------------------------------------------------
// CDNA5 (gfx1250) implementation of a deformable-DETR decoder layer.
// All GEMMs use v_wmma_f32_16x16x32_f16 (f16 operands, f32 accumulate).
// Attention is a flash-style single pass (no 900x900 score materialization).
// Global->LDS staging of contiguous tiles uses the CDNA5 async copy path
// (global_load_async_to_lds_b128 + s_wait_asynccnt).
// ---------------------------------------------------------------------------

typedef __attribute__((ext_vector_type(16))) _Float16 v16h;
typedef __attribute__((ext_vector_type(8)))  float    v8f;

#define NTOK   14400      // LQ*B = 900*16 tokens
#define CDIM   256
#define BATCH  16
#define LQDIM  900
#define LQP    928        // padded query/key length (29*32)
#define NCHUNK 29
#define NQT    57         // ceil(900/16)

union U16 {
  v16h     v;
  unsigned u[8];
  _Float16 h[16];
};

__device__ __forceinline__ v8f wmma16(v16h a, v16h b, v8f c) {
  return __builtin_amdgcn_wmma_f32_16x16x32_f16(false, a, false, b, (short)0, c,
                                                false, false);
}

// CDNA5 async global->LDS copy (ASYNCcnt-tracked). lds_off is the byte offset
// within the workgroup's LDS (low 32 bits of the generic shared address).
__device__ __forceinline__ void async_ld_b128(unsigned lds_off, const void* gptr) {
  asm volatile("global_load_async_to_lds_b128 %0, %1, off"
               :: "v"(lds_off), "v"(gptr)
               : "memory");
}
__device__ __forceinline__ void wait_async0() {
  asm volatile("s_wait_asynccnt 0x0" ::: "memory");
}
__device__ __forceinline__ unsigned lds_offset(const void* p) {
  return (unsigned)(unsigned long long)p;   // flat LDS aperture: addr[31:0]
}

// ---------------------------------------------------------------------------
// Elementwise helpers
// ---------------------------------------------------------------------------
__global__ void cvt_f16_k(const float* __restrict__ s, _Float16* __restrict__ d, int n) {
  int i = blockIdx.x * 256 + threadIdx.x;
  if (i < n) d[i] = (_Float16)s[i];
}

__global__ void qprep_k(const float* __restrict__ qo, const float* __restrict__ qp,
                        _Float16* __restrict__ qoh, _Float16* __restrict__ qsh, int n) {
  int i = blockIdx.x * 256 + threadIdx.x;
  if (i < n) {
    float a = qo[i];
    qoh[i] = (_Float16)a;
    qsh[i] = (_Float16)(a + qp[i]);
  }
}

__global__ void addcvt_k(const float* __restrict__ a, const float* __restrict__ b,
                         _Float16* __restrict__ d, int n) {
  int i = blockIdx.x * 256 + threadIdx.x;
  if (i < n) d[i] = (_Float16)(a[i] + b[i]);
}

// ---------------------------------------------------------------------------
// Generic WMMA GEMM: out = A(f16)[rows,Kd] x W(f16)[Kd(ldw), cols] + bias
//                    (+residual) (ReLU) -> outF(f32) and/or outH(f16)
// grid = (ceil(rows/64), cols/64), block = 256 (8 waves)
// A tile staged via async global->LDS b128; W tile staged with an in-register
// transpose (so B fragment dword pairs are contiguous in LDS).
// ---------------------------------------------------------------------------
__global__ __launch_bounds__(256) void gemm_f16_k(
    const _Float16* __restrict__ A, int Arows, int Kdim,
    const _Float16* __restrict__ W, int ldw,
    const float* __restrict__ bias, const float* __restrict__ residual,
    float* __restrict__ outF, _Float16* __restrict__ outH, int ldo, int relu) {
  __shared__ _Float16 As[64][32];   // A tile, row-major
  __shared__ _Float16 Bs[64][32];   // W tile, transposed: Bs[col][k]

  const int tid  = threadIdx.x;
  const int wave = tid >> 5, lane = tid & 31;
  const int wm = wave & 3, wn = wave >> 2;      // 4x2 wave grid of 16x32 outputs
  const int rowbase = blockIdx.x * 64;
  const int colbase = blockIdx.y * 64;
  const int qrowA = rowbase + (tid >> 2);
  const int ldrow = (qrowA < Arows) ? qrowA : (Arows - 1);
  const int ldcA  = (tid & 3) * 8;
  const int ldkB  = tid >> 3;
  const int ldcB  = (tid & 7) * 8;
  const unsigned a_lds = lds_offset(&As[tid >> 2][ldcA]);

  v8f acc0 = {}; v8f acc1 = {};
  const int nk = Kdim >> 5;
  for (int kt = 0; kt < nk; ++kt) {
    const int k0 = kt << 5;
    // stage A: one async b128 per thread, written directly into LDS
    async_ld_b128(a_lds, A + (size_t)ldrow * Kdim + k0 + ldcA);
    // stage W transposed (coalesced read, scattered LDS write)
    {
      float4 dreg = *(const float4*)(W + (size_t)(k0 + ldkB) * ldw + colbase + ldcB);
      const _Float16* hh = (const _Float16*)&dreg;
#pragma unroll
      for (int j = 0; j < 8; ++j) Bs[ldcB + j][ldkB] = hh[j];
    }
    if (kt + 1 < nk) {  // CDNA5 global_prefetch_b8 for next tiles
      __builtin_prefetch((const void*)(A + (size_t)ldrow * Kdim + k0 + 32 + ldcA), 0, 1);
      __builtin_prefetch((const void*)(W + (size_t)(k0 + 32 + ldkB) * ldw + colbase + ldcB), 0, 1);
    }
    wait_async0();
    __syncthreads();

    // build fragments in contiguous k = 16*hlf + s order (same in A and B)
    U16 af, bf0, bf1;
    const int mrow = wm * 16 + (lane & 15);
    const int hlf  = lane >> 4;
#pragma unroll
    for (int vi = 0; vi < 8; ++vi) {
      int kk = 16 * hlf + 2 * vi;
      af.u[vi] = *(const unsigned*)&As[mrow][kk];
    }
    const int nc = wn * 32 + (lane & 15);
#pragma unroll
    for (int vi = 0; vi < 8; ++vi) {
      int kk = 16 * hlf + 2 * vi;
      bf0.u[vi] = *(const unsigned*)&Bs[nc][kk];
      bf1.u[vi] = *(const unsigned*)&Bs[nc + 16][kk];
    }
    acc0 = wmma16(af.v, bf0.v, acc0);
    acc1 = wmma16(af.v, bf1.v, acc1);
    __syncthreads();
  }

  // epilogue: D layout -> col = l%16, row = r + 8*(l/16)
  const int colA = colbase + wn * 32 + (lane & 15);
  const int colB = colA + 16;
  const float bA = bias[colA], bB = bias[colB];
#pragma unroll
  for (int r = 0; r < 8; ++r) {
    int row = rowbase + wm * 16 + r + 8 * (lane >> 4);
    if (row < Arows) {
      float v0 = acc0[r] + bA;
      float v1 = acc1[r] + bB;
      if (residual) {
        v0 += residual[(size_t)row * ldo + colA];
        v1 += residual[(size_t)row * ldo + colB];
      }
      if (relu) { v0 = fmaxf(v0, 0.f); v1 = fmaxf(v1, 0.f); }
      if (outF) {
        outF[(size_t)row * ldo + colA] = v0;
        outF[(size_t)row * ldo + colB] = v1;
      }
      if (outH) {
        outH[(size_t)row * ldo + colA] = (_Float16)v0;
        outH[(size_t)row * ldo + colB] = (_Float16)v1;
      }
    }
  }
}

// ---------------------------------------------------------------------------
// Repack Q/K/V into per-head zero-padded layout [B][M][LQP][32] (f16),
// scaling Q by CV^-0.5.
// ---------------------------------------------------------------------------
__global__ void repack_k(const _Float16* __restrict__ qk, const _Float16* __restrict__ vv,
                         _Float16* __restrict__ Qb, _Float16* __restrict__ Kb,
                         _Float16* __restrict__ Vb) {
  int t = blockIdx.x * 256 + threadIdx.x;
  const int total = BATCH * 8 * LQP * 32;
  if (t >= total) return;
  int cv = t & 31;
  int r  = t >> 5;
  int q  = r % LQP; r /= LQP;
  int m  = r & 7;
  int b  = r >> 3;
  size_t dst = (((size_t)(b * 8 + m)) * LQP + q) * 32 + cv;
  _Float16 qv = (_Float16)0.f, kv = (_Float16)0.f, xv = (_Float16)0.f;
  if (q < LQDIM) {
    size_t n = (size_t)q * BATCH + b;
    qv = (_Float16)((float)qk[n * 512 + m * 32 + cv] * 0.17677669529663687f);
    kv = qk[n * 512 + 256 + m * 32 + cv];
    xv = vv[n * 256 + m * 32 + cv];
  }
  Qb[dst] = qv; Kb[dst] = kv; Vb[dst] = xv;
}

// ---------------------------------------------------------------------------
// Flash-style attention. One wave per (query-tile, head, batch).
// S^T = K x Q^T (2 WMMA / 32-key chunk), online softmax per query column,
// O^T = V^T x P^T (2 WMMA / chunk).  Output -> [N,256] f16.
// V chunks staged to LDS with async b128 copies.
// ---------------------------------------------------------------------------
__global__ __launch_bounds__(32) void attn_k(const _Float16* __restrict__ Qb,
                                             const _Float16* __restrict__ Kb,
                                             const _Float16* __restrict__ Vb,
                                             _Float16* __restrict__ outH) {
  __shared__ _Float16 Vs[32][32];
  const int lane = threadIdx.x;
  const int qt = blockIdx.x, m = blockIdx.y, b = blockIdx.z;
  const size_t base = ((size_t)(b * 8 + m)) * LQP * 32;
  const int qloc = lane & 15, hlf = lane >> 4;
  const unsigned v_lds = lds_offset(&Vs[lane][0]);

  U16 qf;
  {
    const int qrow = qt * 16 + qloc;
#pragma unroll
    for (int vi = 0; vi < 8; ++vi) {
      int cv = 16 * hlf + 2 * vi;
      qf.u[vi] = *(const unsigned*)(Qb + base + (size_t)qrow * 32 + cv);
    }
  }

  v8f o0 = {}; v8f o1 = {};
  float mrun = -1e30f, lsum = 0.f;

  for (int ch = 0; ch < NCHUNK; ++ch) {
    const int kb = ch * 32;
    // kick off async stage of this chunk's V rows (64B per lane = 4 x b128)
#pragma unroll
    for (int i = 0; i < 4; ++i)
      async_ld_b128(v_lds + 16 * i, Vb + base + (size_t)(kb + lane) * 32 + 8 * i);

    U16 ka0, ka1;
#pragma unroll
    for (int vi = 0; vi < 8; ++vi) {
      int cv = 16 * hlf + 2 * vi;
      ka0.u[vi] = *(const unsigned*)(Kb + base + (size_t)(kb + qloc) * 32 + cv);
      ka1.u[vi] = *(const unsigned*)(Kb + base + (size_t)(kb + 16 + qloc) * 32 + cv);
    }
    v8f s0 = {}, s1 = {};
    s0 = wmma16(ka0.v, qf.v, s0);   // 16 keys x 16 queries
    s1 = wmma16(ka1.v, qf.v, s1);

    // mask padded keys
#pragma unroll
    for (int r = 0; r < 8; ++r) {
      int k0 = kb + r + 8 * hlf;
      if (k0 >= LQDIM)      s0[r] = -1e30f;
      if (k0 + 16 >= LQDIM) s1[r] = -1e30f;
    }
    // online softmax over keys (per query column)
    float mloc = -1e30f;
#pragma unroll
    for (int r = 0; r < 8; ++r) mloc = fmaxf(mloc, fmaxf(s0[r], s1[r]));
    mloc = fmaxf(mloc, __shfl_xor(mloc, 16, 32));
    float mnew = fmaxf(mrun, mloc);
    float corr = __expf(mrun - mnew);
    lsum *= corr;
#pragma unroll
    for (int r = 0; r < 8; ++r) { o0[r] *= corr; o1[r] *= corr; }
    float p0[8], p1[8], ps = 0.f;
#pragma unroll
    for (int r = 0; r < 8; ++r) {
      p0[r] = __expf(s0[r] - mnew);
      p1[r] = __expf(s1[r] - mnew);
      ps += p0[r] + p1[r];
    }
    ps += __shfl_xor(ps, 16, 32);
    lsum += ps;
    mrun = mnew;

    // rebuild P^T (keys x queries) as a B fragment via lane shuffles
    U16 pb;
#pragma unroll
    for (int vi = 0; vi < 8; ++vi) {
      int src = qloc + ((vi >= 4) ? 16 : 0);
      const int r0 = (2 * vi) & 7, r1 = (2 * vi + 1) & 7;
      float a00 = __shfl(p0[r0], src, 32);
      float a10 = __shfl(p1[r0], src, 32);
      float a01 = __shfl(p0[r1], src, 32);
      float a11 = __shfl(p1[r1], src, 32);
      pb.h[2 * vi]     = (_Float16)(hlf ? a10 : a00);
      pb.h[2 * vi + 1] = (_Float16)(hlf ? a11 : a01);
    }

    // drain async V stage, then scattered V^T fragment reads
    wait_async0();
    __syncthreads();
    U16 av0, av1;
#pragma unroll
    for (int vi = 0; vi < 8; ++vi) {
      int kk = 16 * hlf + 2 * vi;
      av0.h[2 * vi]     = Vs[kk][qloc];
      av0.h[2 * vi + 1] = Vs[kk + 1][qloc];
      av1.h[2 * vi]     = Vs[kk][16 + qloc];
      av1.h[2 * vi + 1] = Vs[kk + 1][16 + qloc];
    }
    o0 = wmma16(av0.v, pb.v, o0);   // cv 0..15  x queries
    o1 = wmma16(av1.v, pb.v, o1);   // cv 16..31 x queries
    __syncthreads();
  }

  const float inv = 1.f / lsum;
  const int qrow = qt * 16 + qloc;
  if (qrow < LQDIM) {
    size_t n  = (size_t)qrow * BATCH + b;
    size_t ob = n * 256 + m * 32;
#pragma unroll
    for (int r = 0; r < 8; ++r) {
      int cv0 = r + 8 * hlf;
      outH[ob + cv0]      = (_Float16)(o0[r] * inv);
      outH[ob + 16 + cv0] = (_Float16)(o1[r] * inv);
    }
  }
}

// ---------------------------------------------------------------------------
// LayerNorm over C=256, one row per 256-thread block
// ---------------------------------------------------------------------------
__global__ __launch_bounds__(256) void ln_k(const float* __restrict__ x,
                                            const float* __restrict__ g,
                                            const float* __restrict__ bt,
                                            float* __restrict__ outF,
                                            _Float16* __restrict__ outH) {
  __shared__ float red[256];
  const int row = blockIdx.x, tid = threadIdx.x;
  float v = x[(size_t)row * 256 + tid];
  red[tid] = v; __syncthreads();
  for (int s = 128; s > 0; s >>= 1) { if (tid < s) red[tid] += red[tid + s]; __syncthreads(); }
  float mean = red[0] * (1.f / 256.f);
  __syncthreads();
  float d = v - mean;
  red[tid] = d * d; __syncthreads();
  for (int s = 128; s > 0; s >>= 1) { if (tid < s) red[tid] += red[tid + s]; __syncthreads(); }
  float var = red[0] * (1.f / 256.f);
  float o = d * rsqrtf(var + 1e-5f) * g[tid] + bt[tid];
  if (outF) outF[(size_t)row * 256 + tid] = o;
  if (outH) outH[(size_t)row * 256 + tid] = (_Float16)o;
}

// softmax over L*K=16 attention weights per (token, head)
__global__ void awsm_k(float* __restrict__ aw) {
  int t = blockIdx.x * 256 + threadIdx.x;
  if (t >= NTOK * 8) return;
  float* p = aw + (size_t)(t >> 3) * 128 + (t & 7) * 16;
  float mx = -1e30f, e[16], s = 0.f;
#pragma unroll
  for (int i = 0; i < 16; ++i) mx = fmaxf(mx, p[i]);
#pragma unroll
  for (int i = 0; i < 16; ++i) { e[i] = __expf(p[i] - mx); s += e[i]; }
  float inv = 1.f / s;
#pragma unroll
  for (int i = 0; i < 16; ++i) p[i] = e[i] * inv;
}

// ---------------------------------------------------------------------------
// Deformable sampling: block = (m,cv), one (q,b) per block
// ---------------------------------------------------------------------------
__global__ __launch_bounds__(256) void sample_k(
    const float* __restrict__ ref, const float* __restrict__ off,
    const float* __restrict__ aw,
    const _Float16* __restrict__ vp0, const _Float16* __restrict__ vp1,
    const _Float16* __restrict__ vp2, const _Float16* __restrict__ vp3,
    _Float16* __restrict__ out) {
  const int n = blockIdx.x;                // n = q*B + b
  const int tid = threadIdx.x;
  const int m = tid >> 5, cv = tid & 31;
  const int b = n & 15;
  const float rx = ref[(size_t)n * 2 + 0], ry = ref[(size_t)n * 2 + 1];
  const _Float16* vps[4] = {vp0, vp1, vp2, vp3};
  const int dims[4] = {100, 50, 25, 13};
  float acc = 0.f;
#pragma unroll
  for (int l = 0; l < 4; ++l) {
    const int Wd = dims[l], Hd = dims[l], HW = Wd * Hd;
    const _Float16* vp = vps[l];
#pragma unroll
    for (int k = 0; k < 4; ++k) {
      int oc = ((m * 4 + l) * 4 + k) * 2;
      float ox = off[(size_t)n * 256 + oc];
      float oy = off[(size_t)n * 256 + oc + 1];
      float w8 = aw[(size_t)n * 128 + (m * 4 + l) * 4 + k];
      float px = rx * (float)Wd - 0.5f + ox;
      float py = ry * (float)Hd - 0.5f + oy;
      float x0 = floorf(px), y0 = floorf(py);
      float s = 0.f;
#pragma unroll
      for (int dy = 0; dy < 2; ++dy)
#pragma unroll
        for (int dx = 0; dx < 2; ++dx) {
          float xi = x0 + (float)dx, yi = y0 + (float)dy;
          float wg = (1.f - fabsf(px - xi)) * (1.f - fabsf(py - yi));
          bool valid = (xi >= 0.f) && (xi < (float)Wd) && (yi >= 0.f) && (yi < (float)Hd);
          int xc = (int)fminf(fmaxf(xi, 0.f), (float)(Wd - 1));
          int yc = (int)fminf(fmaxf(yi, 0.f), (float)(Hd - 1));
          size_t idx = ((size_t)b * HW + (size_t)yc * Wd + xc) * 256 + m * 32 + cv;
          float g = (float)vp[idx];
          s += valid ? wg * g : 0.f;
        }
      acc += w8 * s;
    }
  }
  out[(size_t)n * 256 + m * 32 + cv] = (_Float16)acc;
}

// ---------------------------------------------------------------------------
// Host-side orchestration
// ---------------------------------------------------------------------------
extern "C" void kernel_launch(void* const* d_in, const int* in_sizes, int n_in,
                              void* d_out, int out_size, void* d_ws, size_t ws_size,
                              hipStream_t stream) {
  (void)in_sizes; (void)n_in; (void)out_size;
  const float* qo  = (const float*)d_in[0];
  const float* qp  = (const float*)d_in[1];
  const float* ref = (const float*)d_in[2];
  const float* feat[4] = {(const float*)d_in[3], (const float*)d_in[5],
                          (const float*)d_in[7], (const float*)d_in[9]};
  const float* pos[4]  = {(const float*)d_in[4], (const float*)d_in[6],
                          (const float*)d_in[8], (const float*)d_in[10]};
  const float* sa_in_w  = (const float*)d_in[11];
  const float* sa_in_b  = (const float*)d_in[12];
  const float* sa_out_w = (const float*)d_in[13];
  const float* sa_out_b = (const float*)d_in[14];
  const float* n1_g = (const float*)d_in[15]; const float* n1_b = (const float*)d_in[16];
  const float* n2_g = (const float*)d_in[17]; const float* n2_b = (const float*)d_in[18];
  const float* n3_g = (const float*)d_in[19]; const float* n3_b = (const float*)d_in[20];
  const float* val_w = (const float*)d_in[21]; const float* val_b = (const float*)d_in[22];
  const float* off_w = (const float*)d_in[23]; const float* off_b = (const float*)d_in[24];
  const float* aw_w  = (const float*)d_in[25]; const float* aw_b  = (const float*)d_in[26];
  const float* dout_w = (const float*)d_in[27]; const float* dout_b = (const float*)d_in[28];
  const float* ffn_w1 = (const float*)d_in[29]; const float* ffn_b1 = (const float*)d_in[30];
  const float* ffn_w2 = (const float*)d_in[31]; const float* ffn_b2 = (const float*)d_in[32];

  const int HWs[4] = {10000, 2500, 625, 169};
  const int N = NTOK;

  // ---- workspace bump allocator (256B aligned) ----
  char* cur = (char*)d_ws;
  auto take = [&](size_t bytes) -> char* {
    char* r = cur;
    cur += (bytes + 255) & ~(size_t)255;
    return r;
  };
  _Float16* w_in_h   = (_Float16*)take((size_t)256 * 768 * 2);
  _Float16* w_out_h  = (_Float16*)take((size_t)256 * 256 * 2);
  _Float16* w_val_h  = (_Float16*)take((size_t)256 * 256 * 2);
  _Float16* w_off_h  = (_Float16*)take((size_t)256 * 256 * 2);
  _Float16* w_aw_h   = (_Float16*)take((size_t)256 * 128 * 2);
  _Float16* w_dout_h = (_Float16*)take((size_t)256 * 256 * 2);
  _Float16* w_f1_h   = (_Float16*)take((size_t)256 * 1024 * 2);
  _Float16* w_f2_h   = (_Float16*)take((size_t)1024 * 256 * 2);
  _Float16* qo_h = (_Float16*)take((size_t)N * 256 * 2);
  _Float16* qs_h = (_Float16*)take((size_t)N * 256 * 2);
  _Float16* qk_h = (_Float16*)take((size_t)N * 512 * 2);
  _Float16* v_h  = (_Float16*)take((size_t)N * 256 * 2);
  const size_t hsz = (size_t)BATCH * 8 * LQP * 32;
  _Float16* Qb = (_Float16*)take(hsz * 2);
  _Float16* Kb = (_Float16*)take(hsz * 2);
  _Float16* Vb = (_Float16*)take(hsz * 2);
  _Float16* ao_h  = (_Float16*)take((size_t)N * 256 * 2);
  float*    mha_f = (float*)take((size_t)N * 256 * 4);      // reused as dsum
  float*    x1_f  = (float*)take((size_t)N * 256 * 4);
  _Float16* x1_h  = (_Float16*)take((size_t)N * 256 * 2);
  _Float16* fs_h[4]; _Float16* vp_h[4];
  for (int l = 0; l < 4; ++l) fs_h[l] = (_Float16*)take((size_t)BATCH * HWs[l] * 256 * 2);
  for (int l = 0; l < 4; ++l) vp_h[l] = (_Float16*)take((size_t)BATCH * HWs[l] * 256 * 2);
  float* off_f = (float*)take((size_t)N * 256 * 4);         // reused as ffnsum
  float* aw_f  = (float*)take((size_t)N * 128 * 4);
  _Float16* samp_h = (_Float16*)take((size_t)N * 256 * 2);
  float*    x2_f   = (float*)take((size_t)N * 256 * 4);
  _Float16* x2_h   = (_Float16*)take((size_t)N * 256 * 2);
  if ((size_t)(cur - (char*)d_ws) > ws_size) return;        // ws too small
  float*    dsum_f   = mha_f;                               // alias (mha dead after LN1)
  _Float16* ffnh_h   = fs_h[0];                             // alias (featsum dead)
  float*    ffnsum_f = off_f;                               // alias (off dead)

  auto cvt = [&](const float* s, _Float16* d, int n) {
    cvt_f16_k<<<dim3((n + 255) / 256), dim3(256), 0, stream>>>(s, d, n);
  };
  auto gemm = [&](const _Float16* A, int rows, int Kd, const _Float16* W, int ldw,
                  int cols, const float* bias, const float* resid, float* oF,
                  _Float16* oH, int ldo, int relu) {
    dim3 g((rows + 63) / 64, cols / 64);
    gemm_f16_k<<<g, 256, 0, stream>>>(A, rows, Kd, W, ldw, bias, resid, oF, oH, ldo, relu);
  };

  // 1) weights -> f16
  cvt(sa_in_w, w_in_h, 256 * 768);
  cvt(sa_out_w, w_out_h, 256 * 256);
  cvt(val_w, w_val_h, 256 * 256);
  cvt(off_w, w_off_h, 256 * 256);
  cvt(aw_w, w_aw_h, 256 * 128);
  cvt(dout_w, w_dout_h, 256 * 256);
  cvt(ffn_w1, w_f1_h, 256 * 1024);
  cvt(ffn_w2, w_f2_h, 1024 * 256);

  // 2) q = qo+qp (f16) and qo (f16)
  qprep_k<<<dim3(N * 256 / 256), dim3(256), 0, stream>>>(qo, qp, qo_h, qs_h, N * 256);

  // 3) QK = q @ Win[:, :512] + b ; V = qo @ Win[:, 512:768] + b
  gemm(qs_h, N, 256, w_in_h, 768, 512, sa_in_b, nullptr, nullptr, qk_h, 512, 0);
  gemm(qo_h, N, 256, w_in_h + 512, 768, 256, sa_in_b + 512, nullptr, nullptr, v_h, 256, 0);

  // 4) repack per-head padded Q/K/V (Q pre-scaled by CV^-0.5)
  repack_k<<<dim3((int)((hsz + 255) / 256)), dim3(256), 0, stream>>>(qk_h, v_h, Qb, Kb, Vb);

  // 5) flash attention
  attn_k<<<dim3(NQT, 8, BATCH), dim3(32), 0, stream>>>(Qb, Kb, Vb, ao_h);

  // 6) out-proj + residual(query_objects); LN1
  gemm(ao_h, N, 256, w_out_h, 256, 256, sa_out_b, qo, mha_f, nullptr, 256, 0);
  ln_k<<<dim3(N), dim3(256), 0, stream>>>(mha_f, n1_g, n1_b, x1_f, x1_h);

  // 7) feat+pos (f16), value projections per level
  for (int l = 0; l < 4; ++l) {
    int n = BATCH * HWs[l] * 256;
    addcvt_k<<<dim3((n + 255) / 256), dim3(256), 0, stream>>>(feat[l], pos[l], fs_h[l], n);
  }
  for (int l = 0; l < 4; ++l)
    gemm(fs_h[l], BATCH * HWs[l], 256, w_val_h, 256, 256, val_b, nullptr, nullptr,
         vp_h[l], 256, 0);

  // 8) sampling offsets + attention weights (+softmax)
  gemm(x1_h, N, 256, w_off_h, 256, 256, off_b, nullptr, off_f, nullptr, 256, 0);
  gemm(x1_h, N, 256, w_aw_h, 256, 128, aw_b, nullptr, aw_f, nullptr, 128, 0);
  awsm_k<<<dim3((N * 8 + 255) / 256), dim3(256), 0, stream>>>(aw_f);

  // 9) bilinear gather + weighted sum
  sample_k<<<dim3(N), dim3(256), 0, stream>>>(ref, off_f, aw_f, vp_h[0], vp_h[1],
                                              vp_h[2], vp_h[3], samp_h);

  // 10) deform out-proj + residual(x1); LN2
  gemm(samp_h, N, 256, w_dout_h, 256, 256, dout_b, x1_f, dsum_f, nullptr, 256, 0);
  ln_k<<<dim3(N), dim3(256), 0, stream>>>(dsum_f, n2_g, n2_b, x2_f, x2_h);

  // 11) FFN + residual(x2); LN3 -> d_out
  gemm(x2_h, N, 256, w_f1_h, 1024, 1024, ffn_b1, nullptr, nullptr, ffnh_h, 1024, 1);
  gemm(ffnh_h, N, 1024, w_f2_h, 256, 256, ffn_b2, x2_f, ffnsum_f, nullptr, 256, 0);
  ln_k<<<dim3(N), dim3(256), 0, stream>>>(ffnsum_f, n3_g, n3_b, (float*)d_out, nullptr);
}
